// GraphConvolution_91233695302270
// MI455X (gfx1250) — compile-verified, matrix operand order checked by reading, not run on hardware
//
#include <hip/hip_runtime.h>
#include <stdint.h>

// ---- problem dims (from reference) ----
#define NN 512   // nodes
#define BB 64    // batch
#define FF 128   // in features
#define OO 128   // out features
#define EE 3     // edge channels

typedef __attribute__((ext_vector_type(16))) __bf16 v16bf;
typedef __attribute__((ext_vector_type(8)))  float  v8f;

union FragBF {
    v16bf    v;
    uint32_t u[8];
    uint4    q[2];
};

__device__ __forceinline__ uint32_t bf16r(float f) {
    return (__float_as_uint(f) + 0x8000u) >> 16;       // round-to-nearest (no tie fix)
}
__device__ __forceinline__ uint32_t pk_bf16(float lo, float hi) {
    return bf16r(lo) | ((__float_as_uint(hi) + 0x8000u) & 0xffff0000u);
}
__device__ __forceinline__ float bf16_to_f32(uint16_t v) {
    return __uint_as_float((uint32_t)v << 16);
}

// -------------------------------------------------------------------------
// Kernel C: streaming f32 -> bf16 convert (8 elems / thread, 16B stores)
// -------------------------------------------------------------------------
__global__ __launch_bounds__(256)
void cvt_bf16_kernel(const float* __restrict__ src, uint16_t* __restrict__ dst, int n8)
{
    const int i = blockIdx.x * 256 + threadIdx.x;
    if (i >= n8) return;
    const float4* s = (const float4*)src + (size_t)i * 2;
    float4 a = s[0];
    float4 b = s[1];
    uint4 o;
    o.x = pk_bf16(a.x, a.y);
    o.y = pk_bf16(a.z, a.w);
    o.z = pk_bf16(b.x, b.y);
    o.w = pk_bf16(b.z, b.w);
    ((uint4*)dst)[i] = o;
}

// -------------------------------------------------------------------------
// Kernel A: support[e] = X @ W[e] + b[e]  (stored transposed, bf16, [E,B,O,N])
//           t          = sigmoid(X @ Wh + bh)  (f32, [N,B,O])
// X already bf16. grid = (256 row tiles, 4 weights), block = 256 (8 waves)
// -------------------------------------------------------------------------
__global__ __launch_bounds__(256)
void gcn_support_kernel(const uint16_t* __restrict__ xb,   // [N*B, F] bf16
                        const float*    __restrict__ W,    // [E, F, O]
                        const float*    __restrict__ bE,   // [E, 1, O]
                        const float*    __restrict__ Wh,   // [F, O]
                        const float*    __restrict__ bh,   // [1, O]
                        uint16_t*       __restrict__ ST,   // [E, B, O, N] bf16
                        float*          __restrict__ T)    // [N*B, O] f32
{
    const int w       = blockIdx.y;            // 0..2 = edge weight, 3 = highway
    const int rowbase = blockIdx.x * 128;
    const int tid     = threadIdx.x;
    const int lane    = tid & 31;              // wave32
    const int wave    = tid >> 5;

    const float* Wsrc = (w < 3) ? (W + (size_t)w * FF * OO) : Wh;
    const float* bias = (w < 3) ? (bE + (size_t)w * OO)     : bh;

    // Transposed bf16 weights in LDS: Wt[o][f], row stride 136 elems (272B, 16B aligned)
    __shared__ __align__(16) uint16_t Wt[OO * 136];
    for (int idx = tid; idx < FF * OO; idx += 256) {
        int f = idx >> 7;
        int o = idx & 127;
        Wt[o * 136 + f] = (uint16_t)bf16r(Wsrc[(size_t)f * OO + o]);
    }
    __syncthreads();

    const int mrow = rowbase + wave * 16 + (lane & 15);
    const int koff = (lane < 16) ? 0 : 8;     // K sub-offset per ISA 16-bit A layout
    const int bsel = (lane < 16) ? 0 : 16;    // byte select for B fragment rows

    v8f acc[8];
#pragma unroll
    for (int j = 0; j < 8; ++j)
#pragma unroll
        for (int q = 0; q < 8; ++q) acc[j][q] = 0.0f;

    for (int kc = 0; kc < FF; kc += 32) {
        // A fragment: xb[mrow][kc+koff..+8) and xb[mrow][kc+16+koff..+8) (already bf16)
        FragBF a;
        const uint16_t* xr = xb + (size_t)mrow * FF + kc + koff;
        a.q[0] = *(const uint4*)xr;
        a.q[1] = *(const uint4*)(xr + 16);
#pragma unroll
        for (int j = 0; j < 8; ++j) {
            FragBF bf;
            const char* brow = (const char*)(Wt + ((j * 16 + (lane & 15)) * 136 + kc));
            bf.q[0] = *(const uint4*)(brow + bsel);
            bf.q[1] = *(const uint4*)(brow + bsel + 32);
            acc[j] = __builtin_amdgcn_wmma_f32_16x16x32_bf16(
                         false, a.v, false, bf.v, (short)0, acc[j], false, false);
        }
    }

    // Epilogue: C/D layout -> lane l: o = tile*16 + (l&15); rows m = base + i + 8*(l>>4)
    const int mbase = rowbase + wave * 16 + ((lane >> 4) << 3);
#pragma unroll
    for (int j = 0; j < 8; ++j) {
        const int   o  = j * 16 + (lane & 15);
        const float bv = bias[o];
        if (w < 3) {
#pragma unroll
            for (int i = 0; i < 8; ++i) {
                const int m = mbase + i;
                const int n = m >> 6;      // node index
                const int b = m & 63;      // batch index
                const float v = acc[j][i] + bv;
                ST[(((size_t)w * BB + b) * OO + o) * NN + n] = (uint16_t)bf16r(v);
            }
        } else {
#pragma unroll
            for (int i = 0; i < 8; ++i) {
                const int m = mbase + i;
                const float v = acc[j][i] + bv;
                T[(size_t)m * OO + o] = 1.0f / (1.0f + __expf(-v));
            }
        }
    }
}

// -------------------------------------------------------------------------
// Kernel B: out[m,b,:] = relu( sum_e adj[e,b] @ support[e,b] )
//           h = out * t + x * (1 - t)
// adj already bf16 (L2-resident across both layers: 100MB < 192MB L2).
// grid = (4 m-tiles, 64 batches), block = 256 (8 waves, 16 rows each)
// -------------------------------------------------------------------------
__global__ __launch_bounds__(256)
void gcn_aggregate_kernel(const uint16_t* __restrict__ adjb,  // [E,B,N,N] bf16
                          const uint16_t* __restrict__ ST,    // [E,B,O,N] bf16
                          const float*    __restrict__ T,     // [N*B, O] f32
                          const float*    __restrict__ xinF,  // [N*B, O] f32 (layer 1) or null
                          const uint16_t* __restrict__ xinB,  // [N*B, O] bf16 (layer 2) or null
                          float*          __restrict__ houtF, // f32 out (layer 2) or null
                          uint16_t*       __restrict__ houtB) // bf16 out (layer 1) or null
{
    const int b       = blockIdx.y;
    const int rowbase = blockIdx.x * 128;
    const int tid     = threadIdx.x;
    const int lane    = tid & 31;
    const int wave    = tid >> 5;

    // B-operand tile: supportT rows [o][k0..k0+32), 64B per row -> 8KB
    __shared__ __align__(16) uint16_t Btile[OO * 32];

    const int mrow = rowbase + wave * 16 + (lane & 15);
    const int koff = (lane < 16) ? 0 : 8;
    const int bsel = (lane < 16) ? 0 : 16;

    v8f acc[8];
#pragma unroll
    for (int j = 0; j < 8; ++j)
#pragma unroll
        for (int q = 0; q < 8; ++q) acc[j][q] = 0.0f;

    const int so    = tid >> 1;  // staged o-row per thread
    const int shalf = tid & 1;   // 16-element half

    for (int e = 0; e < EE; ++e) {
        const uint16_t* adjrow = adjb + (((size_t)e * BB + b) * NN + mrow) * NN;
        const uint16_t* stb    = ST + ((size_t)e * BB + b) * OO * NN;
        for (int k0 = 0; k0 < NN; k0 += 32) {
            __syncthreads();   // previous tile fully consumed
            {   // stage 32 K-columns of supportT for all 128 o-rows
                const uint4* src = (const uint4*)(stb + (size_t)so * NN + k0 + shalf * 16);
                uint4 v0 = src[0];
                uint4 v1 = src[1];
                uint4* dst = (uint4*)(Btile + so * 32 + shalf * 16);
                dst[0] = v0;
                dst[1] = v1;
            }
            __syncthreads();

            if (k0 + 32 < NN)   // pull next adj chunk toward the caches
                __builtin_prefetch(adjrow + k0 + 32, 0, 1);

            // A fragment: adj bf16 row, already in ISA 16-bit A-operand layout
            FragBF a;
            const uint16_t* ar = adjrow + k0 + koff;
            a.q[0] = *(const uint4*)ar;          // K koff..koff+7
            a.q[1] = *(const uint4*)(ar + 16);   // K koff+16..koff+23
#pragma unroll
            for (int j = 0; j < 8; ++j) {
                FragBF bf;
                const char* brow = (const char*)(Btile + (j * 16 + (lane & 15)) * 32);
                bf.q[0] = *(const uint4*)(brow + bsel);
                bf.q[1] = *(const uint4*)(brow + bsel + 32);
                acc[j] = __builtin_amdgcn_wmma_f32_16x16x32_bf16(
                             false, a.v, false, bf.v, (short)0, acc[j], false, false);
            }
        }
    }

    // Epilogue: ReLU + highway gating, fused
    const int mbase = rowbase + wave * 16 + ((lane >> 4) << 3);
#pragma unroll
    for (int j = 0; j < 8; ++j) {
        const int o = j * 16 + (lane & 15);
#pragma unroll
        for (int i = 0; i < 8; ++i) {
            const int m = mbase + i;
            const size_t idx = ((size_t)m * BB + b) * OO + o;
            float v = acc[j][i];
            v = v > 0.0f ? v : 0.0f;
            const float t  = T[idx];
            const float xi = xinF ? xinF[idx] : bf16_to_f32(xinB[idx]);
            const float h  = v * t + xi * (1.0f - t);
            if (houtF) houtF[idx] = h;
            if (houtB) houtB[idx] = (uint16_t)bf16r(h);
        }
    }
}

// -------------------------------------------------------------------------
extern "C" void kernel_launch(void* const* d_in, const int* in_sizes, int n_in,
                              void* d_out, int out_size, void* d_ws, size_t ws_size,
                              hipStream_t stream) {
    const float* x   = (const float*)d_in[0];  // [N,B,F]
    const float* adj = (const float*)d_in[1];  // [E,B,N,N]
    const float* W1  = (const float*)d_in[2];
    const float* b1  = (const float*)d_in[3];
    const float* Wh1 = (const float*)d_in[4];
    const float* bh1 = (const float*)d_in[5];
    const float* W2  = (const float*)d_in[6];
    const float* b2  = (const float*)d_in[7];
    const float* Wh2 = (const float*)d_in[8];
    const float* bh2 = (const float*)d_in[9];
    float* out = (float*)d_out;

    // workspace layout
    char* ws = (char*)d_ws;
    const size_t ADJ_ELEMS = (size_t)EE * BB * NN * NN;  // 50,331,648
    const size_t X_ELEMS   = (size_t)NN * BB * FF;       //  4,194,304
    uint16_t* ADJB = (uint16_t*)ws;                                       // 100,663,296 B
    uint16_t* XB   = (uint16_t*)(ws + ADJ_ELEMS * 2);                     //   8,388,608 B
    uint16_t* ST   = (uint16_t*)(ws + ADJ_ELEMS * 2 + X_ELEMS * 2);       //  25,165,824 B
    float*    T    = (float*)((char*)ST + (size_t)EE * BB * OO * NN * 2); //  16,777,216 B
    uint16_t* H1B  = (uint16_t*)((char*)T + (size_t)NN * BB * OO * 4);    //   8,388,608 B

    dim3 blk(256);
    dim3 gA((NN * BB) / 128, 4);   // 256 row-tiles x 4 weight matrices
    dim3 gB(NN / 128, BB);         // 4 m-tiles x 64 batches

    // one-shot bf16 conversion of the streamed operands
    const int adj8 = (int)(ADJ_ELEMS / 8);
    const int x8   = (int)(X_ELEMS / 8);
    cvt_bf16_kernel<<<(adj8 + 255) / 256, blk, 0, stream>>>(adj, ADJB, adj8);
    cvt_bf16_kernel<<<(x8 + 255) / 256, blk, 0, stream>>>(x, XB, x8);

    // layer 1 (gating carry uses f32 x; h1 kept as bf16 only)
    gcn_support_kernel<<<gA, blk, 0, stream>>>(XB, W1, b1, Wh1, bh1, ST, T);
    gcn_aggregate_kernel<<<gB, blk, 0, stream>>>(ADJB, ST, T, x, nullptr, nullptr, H1B);
    // layer 2 (adj_bf16 should still be L2-resident)
    gcn_support_kernel<<<gA, blk, 0, stream>>>(H1B, W2, b2, Wh2, bh2, ST, T);
    gcn_aggregate_kernel<<<gB, blk, 0, stream>>>(ADJB, ST, T, nullptr, H1B, out, nullptr);
}
